// Encoder_33603824123920
// MI455X (gfx1250) — compile-verified, matrix operand order checked by reading, not run on hardware
//
#include <hip/hip_runtime.h>

// ---------------------------------------------------------------------------
// Encoder layer forward for MI455X (gfx1250, wave32, WMMA).
// B=2 S=2048 D=1024 H=16 hd=64 d_ff=4096.  All matmuls in bf16 WMMA, f32 acc.
// ---------------------------------------------------------------------------

#define DEV __device__ __forceinline__

typedef __attribute__((ext_vector_type(16))) __bf16 v16bf;
typedef __attribute__((ext_vector_type(8)))  float  v8f;

union Frag16 {
    v16bf v;
    uint4 u[2];
    unsigned short s[16];
};

DEV unsigned short f2bf(float f) {   // round-to-nearest-even f32 -> bf16 bits
    unsigned int u = __float_as_uint(f);
    u += 0x7FFFu + ((u >> 16) & 1u);
    return (unsigned short)(u >> 16);
}

static constexpr int Bb   = 2;
static constexpr int S    = 2048;
static constexpr int Dd   = 1024;
static constexpr int Hh   = 16;
static constexpr int HD   = 64;
static constexpr int DFF  = 4096;
static constexpr int MTOK = Bb * S;          // 4096 tokens

DEV v8f wmma_bf16(const v16bf& a, const v16bf& b, const v8f& c) {
    return __builtin_amdgcn_wmma_f32_16x16x32_bf16(
        /*neg_a=*/false, a, /*neg_b=*/false, b,
        /*c_mod=*/(short)0, c, /*reuse_a=*/false, /*reuse_b=*/false);
}

// ------------------------------ embedding ----------------------------------
__global__ void k_embed(const int* __restrict__ tok, const float* __restrict__ emb,
                        float* __restrict__ x, unsigned short* __restrict__ xb) {
    int idx = blockIdx.x * blockDim.x + threadIdx.x;     // over MTOK*D = 4M
    int t = idx >> 10;                                    // / D
    int d = idx & (Dd - 1);
    float v = emb[(size_t)tok[t] * Dd + d];
    x[idx]  = v;
    xb[idx] = f2bf(v);
}

// --------------- per-head transpose x -> xT[b,h][hd][S] (bf16) -------------
__global__ void k_xT(const float* __restrict__ x, unsigned short* __restrict__ xT) {
    int idx  = blockIdx.x * blockDim.x + threadIdx.x;    // over B*H*HD*S = 4M
    int s    = idx & (S - 1);
    int rest = idx >> 11;                                 // / S
    int d    = rest & (HD - 1);
    int bh   = rest >> 6;
    int b    = bh >> 4, h = bh & 15;
    xT[idx] = f2bf(x[((size_t)(b * S + s) << 10) + h * HD + d]);
}

// ------------------- weight convert+transpose: W[K x N] -> Wt[N x K] -------
__global__ void k_wT(const float* __restrict__ w, unsigned short* __restrict__ wt,
                     int K, int N) {
    int idx = blockIdx.x * blockDim.x + threadIdx.x;     // over N*K
    int k = idx % K;
    int n = idx / K;
    wt[idx] = f2bf(w[(size_t)k * N + n]);
}

// --------------------------- WMMA GEMM -------------------------------------
// C[M x N] = act(A[M x K] * Bt^T + bias).  A,Bt row-major bf16 (Bt = W^T).
// Block: 128 thr (4 waves) -> 128x64 tile; wave -> 64x32 (4x2 fragments).
template <bool HAS_BIAS, bool RELU, bool OUT_BF16>
__global__ void __launch_bounds__(128)
k_gemm(const unsigned short* __restrict__ A, const unsigned short* __restrict__ Bt,
       const float* __restrict__ bias, float* __restrict__ Cf,
       unsigned short* __restrict__ Cb, int Ndim, int K) {
    const int lane = threadIdx.x & 31;
    const int wave = threadIdx.x >> 5;
    const int lo = lane & 15, hi = lane >> 4;
    const int m0 = blockIdx.y * 128 + (wave >> 1) * 64;
    const int n0 = blockIdx.x * 64  + (wave & 1)  * 32;

    v8f acc[4][2];
#pragma unroll
    for (int i = 0; i < 4; ++i)
#pragma unroll
        for (int j = 0; j < 2; ++j) acc[i][j] = (v8f){0,0,0,0,0,0,0,0};

    for (int kk = 0; kk < K; kk += 32) {
        Frag16 af[4], bfr[2];
#pragma unroll
        for (int i = 0; i < 4; ++i) {
            const unsigned short* p = A + (size_t)(m0 + i * 16 + lo) * K + kk + hi * 8;
            af[i].u[0] = *(const uint4*)p;
            af[i].u[1] = *(const uint4*)(p + 16);
        }
#pragma unroll
        for (int j = 0; j < 2; ++j) {
            const unsigned short* p = Bt + (size_t)(n0 + j * 16 + lo) * K + kk + hi * 8;
            bfr[j].u[0] = *(const uint4*)p;
            bfr[j].u[1] = *(const uint4*)(p + 16);
        }
#pragma unroll
        for (int i = 0; i < 4; ++i)
#pragma unroll
            for (int j = 0; j < 2; ++j)
                acc[i][j] = wmma_bf16(af[i].v, bfr[j].v, acc[i][j]);
    }

#pragma unroll
    for (int i = 0; i < 4; ++i)
#pragma unroll
        for (int j = 0; j < 2; ++j) {
            int col  = n0 + j * 16 + lo;
            float bv = HAS_BIAS ? bias[col] : 0.0f;
#pragma unroll
            for (int r = 0; r < 8; ++r) {
                int row = m0 + i * 16 + hi * 8 + r;   // C-frag row = r + 8*hi
                float v = acc[i][j][r] + bv;
                if (RELU) v = fmaxf(v, 0.0f);
                if (OUT_BF16) Cb[(size_t)row * Ndim + col] = f2bf(v);
                else          Cf[(size_t)row * Ndim + col] = v;
            }
        }
}

// ------------------------ flash attention (1 wave) -------------------------
// One wave: 16 queries of head (b,h), streams 2048 keys in 32-key blocks.
// Scores computed transposed (St = K*Q^T) so exp'd probs are lane-local
// when repacked as the A-fragment of P*V.
__global__ void __launch_bounds__(32)
k_attn(const unsigned short* __restrict__ xb, const unsigned short* __restrict__ xT,
       unsigned short* __restrict__ ctx) {
    const int lane = threadIdx.x;
    const int lo = lane & 15, hi = lane >> 4;
    const int qt = blockIdx.x;            // 0..127 (query tile of 16)
    const int bh = blockIdx.y;            // 0..31
    const int b = bh >> 4, h = bh & 15;

    const unsigned short* xq  = xb + (size_t)(b * S + qt * 16) * Dd + h * HD;
    const unsigned short* xk  = xb + (size_t)b * S * Dd + h * HD;
    const unsigned short* xTb = xT + (size_t)bh * HD * S;

    // Q as B-fragments (lane = query column), hd split into two K=32 chunks
    Frag16 qf[2];
#pragma unroll
    for (int kc = 0; kc < 2; ++kc) {
        const unsigned short* p = xq + (size_t)lo * Dd + kc * 32 + hi * 8;
        qf[kc].u[0] = *(const uint4*)p;
        qf[kc].u[1] = *(const uint4*)(p + 16);
    }

    v8f acc[4];
#pragma unroll
    for (int f = 0; f < 4; ++f) acc[f] = (v8f){0,0,0,0,0,0,0,0};
    float m_run = -1e30f, l_run = 0.0f;
    const float sc = 0.125f;              // 1/sqrt(64)

    for (int kb = 0; kb < S / 32; ++kb) {
        // St[32 keys x 16 queries] as two 16x16 C-frags
        v8f st0 = (v8f){0,0,0,0,0,0,0,0}, st1 = (v8f){0,0,0,0,0,0,0,0};
#pragma unroll
        for (int kc = 0; kc < 2; ++kc) {
            Frag16 kf0, kf1;
            const unsigned short* p0 = xk + (size_t)(kb * 32 + lo) * Dd      + kc * 32 + hi * 8;
            const unsigned short* p1 = xk + (size_t)(kb * 32 + 16 + lo) * Dd + kc * 32 + hi * 8;
            kf0.u[0] = *(const uint4*)p0; kf0.u[1] = *(const uint4*)(p0 + 16);
            kf1.u[0] = *(const uint4*)p1; kf1.u[1] = *(const uint4*)(p1 + 16);
            st0 = wmma_bf16(kf0.v, qf[kc].v, st0);
            st1 = wmma_bf16(kf1.v, qf[kc].v, st1);
        }

        // online softmax; lane owns query `lo`, partner (xor 16) has the
        // complementary 16 keys of this block
        float pv[16];
#pragma unroll
        for (int r = 0; r < 8; ++r) { pv[r] = st0[r] * sc; pv[8 + r] = st1[r] * sc; }
        float mloc = -1e30f;
#pragma unroll
        for (int e = 0; e < 16; ++e) mloc = fmaxf(mloc, pv[e]);
        mloc = fmaxf(mloc, __shfl_xor(mloc, 16, 32));
        float m_new = fmaxf(m_run, mloc);
        float ssum = 0.0f;
#pragma unroll
        for (int e = 0; e < 16; ++e) { pv[e] = __expf(pv[e] - m_new); ssum += pv[e]; }
        ssum += __shfl_xor(ssum, 16, 32);
        float cs = __expf(m_run - m_new);
        l_run = l_run * cs + ssum;
        m_run = m_new;

        // rescale ctx accumulators: row r of a C-frag is query r + 8*hi,
        // whose stats live in lane (r + 8*hi)
        float rs[8];
#pragma unroll
        for (int r = 0; r < 8; ++r) rs[r] = __shfl(cs, r + hi * 8, 32);
#pragma unroll
        for (int f = 0; f < 4; ++f)
#pragma unroll
            for (int r = 0; r < 8; ++r) acc[f][r] *= rs[r];

        // P A-fragment is lane-local: element e == pv[e] (keys kIdx(e,hi))
        Frag16 pf;
#pragma unroll
        for (int e = 0; e < 16; ++e) pf.s[e] = f2bf(pv[e]);

        // ctx += P[16x32] * V[32x64]; V B-frags from transposed xT rows
#pragma unroll
        for (int f = 0; f < 4; ++f) {
            Frag16 vf;
            const unsigned short* p = xTb + (size_t)(f * 16 + lo) * S + kb * 32 + hi * 8;
            vf.u[0] = *(const uint4*)p;
            vf.u[1] = *(const uint4*)(p + 16);
            acc[f] = wmma_bf16(pf.v, vf.v, acc[f]);
        }
    }

    float inv = 1.0f / l_run;
    float ri[8];
#pragma unroll
    for (int r = 0; r < 8; ++r) ri[r] = __shfl(inv, r + hi * 8, 32);
#pragma unroll
    for (int f = 0; f < 4; ++f)
#pragma unroll
        for (int r = 0; r < 8; ++r) {
            int tokrow = qt * 16 + hi * 8 + r;
            ctx[(size_t)(b * S + tokrow) * Dd + h * HD + f * 16 + lo] =
                f2bf(acc[f][r] * ri[r]);
        }
}

// ----------------------- residual add + LayerNorm --------------------------
template <bool WRITE_BF16>
__global__ void __launch_bounds__(256)
k_ln(const float* __restrict__ a, const float* __restrict__ res,
     const float* __restrict__ gamma, const float* __restrict__ beta,
     float* __restrict__ of, unsigned short* __restrict__ ob) {
    int t = blockIdx.x, tid = threadIdx.x;
    const float* pa = a   + (size_t)t * Dd;
    const float* pr = res + (size_t)t * Dd;
    float v[4], s = 0.0f, s2 = 0.0f;
#pragma unroll
    for (int i = 0; i < 4; ++i) {
        int d = tid + i * 256;
        float x = pa[d] + pr[d];
        v[i] = x; s += x; s2 += x * x;
    }
    __shared__ float red[256], red2[256];
    red[tid] = s; red2[tid] = s2;
    __syncthreads();
    for (int off = 128; off > 0; off >>= 1) {
        if (tid < off) { red[tid] += red[tid + off]; red2[tid] += red2[tid + off]; }
        __syncthreads();
    }
    float mean = red[0]  * (1.0f / Dd);
    float var  = red2[0] * (1.0f / Dd) - mean * mean;
    float rstd = rsqrtf(var + 1e-5f);
#pragma unroll
    for (int i = 0; i < 4; ++i) {
        int d = tid + i * 256;
        float y = (v[i] - mean) * rstd * gamma[d] + beta[d];
        of[(size_t)t * Dd + d] = y;
        if (WRITE_BF16) ob[(size_t)t * Dd + d] = f2bf(y);
    }
}

// ---------------------------------------------------------------------------
extern "C" void kernel_launch(void* const* d_in, const int* in_sizes, int n_in,
                              void* d_out, int out_size, void* d_ws, size_t ws_size,
                              hipStream_t stream) {
    (void)in_sizes; (void)n_in; (void)out_size; (void)ws_size;
    const int*   tok = (const int*)d_in[0];
    const float* emb = (const float*)d_in[2];
    const float* w_o = (const float*)d_in[3];
    const float* w1  = (const float*)d_in[4];
    const float* b1  = (const float*)d_in[5];
    const float* w2  = (const float*)d_in[6];
    const float* b2  = (const float*)d_in[7];
    const float* g1  = (const float*)d_in[8];
    const float* be1 = (const float*)d_in[9];
    const float* g2  = (const float*)d_in[10];
    const float* be2 = (const float*)d_in[11];
    float* out = (float*)d_out;

    char* ws = (char*)d_ws;
    size_t off = 0;
    auto alloc = [&](size_t bytes) -> void* {
        void* p = ws + off;
        off = (off + bytes + 255) & ~(size_t)255;
        return p;
    };
    float*          x        = (float*)alloc((size_t)MTOK * Dd * 4);
    unsigned short* xb       = (unsigned short*)alloc((size_t)MTOK * Dd * 2);
    unsigned short* xT       = (unsigned short*)alloc((size_t)Bb * Hh * HD * S * 2);
    unsigned short* ctxb     = (unsigned short*)alloc((size_t)MTOK * Dd * 2);
    unsigned short* woT      = (unsigned short*)alloc((size_t)Dd * Dd * 2);
    unsigned short* w1T      = (unsigned short*)alloc((size_t)Dd * DFF * 2);
    unsigned short* w2T      = (unsigned short*)alloc((size_t)DFF * Dd * 2);
    float*          attn_out = (float*)alloc((size_t)MTOK * Dd * 4);
    float*          x1       = (float*)alloc((size_t)MTOK * Dd * 4);
    unsigned short* x1b      = (unsigned short*)alloc((size_t)MTOK * Dd * 2);
    unsigned short* hb       = (unsigned short*)alloc((size_t)MTOK * DFF * 2);
    float*          ff       = (float*)alloc((size_t)MTOK * Dd * 4);

    // 1) embedding gather (f32 + bf16 copies)
    k_embed<<<(MTOK * Dd) / 256, 256, 0, stream>>>(tok, emb, x, xb);
    // 2) per-head transpose for attention V/K streaming
    k_xT<<<(Bb * Hh * HD * S) / 256, 256, 0, stream>>>(x, xT);
    // 3) bf16 weight transposes
    k_wT<<<(Dd * Dd) / 256, 256, 0, stream>>>(w_o, woT, Dd, Dd);
    k_wT<<<(Dd * DFF) / 256, 256, 0, stream>>>(w1, w1T, Dd, DFF);
    k_wT<<<(DFF * Dd) / 256, 256, 0, stream>>>(w2, w2T, DFF, Dd);
    // 4) flash attention -> ctx (bf16)
    k_attn<<<dim3(S / 16, Bb * Hh), 32, 0, stream>>>(xb, xT, ctxb);
    // 5) attn_out = ctx @ w_o
    k_gemm<false, false, false><<<dim3(Dd / 64, MTOK / 128), 128, 0, stream>>>(
        ctxb, woT, nullptr, attn_out, nullptr, Dd, Dd);
    // 6) x1 = LN(x + attn_out)
    k_ln<true><<<MTOK, 256, 0, stream>>>(x, attn_out, g1, be1, x1, x1b);
    // 7) h = relu(x1 @ w1 + b1)  (bf16)
    k_gemm<true, true, true><<<dim3(DFF / 64, MTOK / 128), 128, 0, stream>>>(
        x1b, w1T, b1, nullptr, hb, DFF, Dd);
    // 8) ff = h @ w2 + b2
    k_gemm<true, false, false><<<dim3(Dd / 64, MTOK / 128), 128, 0, stream>>>(
        hb, w2T, b2, ff, nullptr, Dd, DFF);
    // 9) out = LN(x1 + ff)
    k_ln<false><<<MTOK, 256, 0, stream>>>(x1, ff, g2, be2, out, nullptr);
}
